// TransformerDecoder_2800318677660
// MI455X (gfx1250) — matrix operand order repaired and checked by reading, not compiled, not run on hardware
//
#include <hip/hip_runtime.h>

typedef __bf16 bf16;
typedef __attribute__((ext_vector_type(16))) __bf16 v16bf;
typedef __attribute__((ext_vector_type(8)))  float  v8f;
typedef __attribute__((ext_vector_type(4)))  int    v4i;

#define DM    1024
#define DFF_N 4096
#define NH    16
#define HDK   64
#define TSEQ  512
#define NB    4
#define NL    4
#define NVOC  32000
#define MTOK  2048   // B*T

#define BM 256       // block M tile (8 waves x 32 rows)
#define BN 64        // block N tile (4 x 16)
#define BK 32        // K step = one WMMA K
#define APAD 40      // LDS row pitch in bf16 (32 + 8 pad, keeps 16B alignment)
#define BPAD 40

// ---------------------------------------------------------------------------
// CDNA5 async global->LDS copy (ASYNCcnt path) with safe fallback.
// Probe result: builtin exists, signature is
//   (v4i __device__ *gsrc, v4i __shared__ *ldst, int offset, int cpol)
// ---------------------------------------------------------------------------
#if defined(__has_builtin)
#if __has_builtin(__builtin_amdgcn_global_load_async_to_lds_b128)
#define HAVE_ASYNC_LDS 1
#endif
#endif

__device__ __forceinline__ void lds_cp16(void *lds_dst, const void *gsrc)
{
#if defined(HAVE_ASYNC_LDS)
  __builtin_amdgcn_global_load_async_to_lds_b128(
      (__attribute__((address_space(1))) v4i *)gsrc,
      (__attribute__((address_space(3))) v4i *)lds_dst, 0, 0);
#else
  *(float4 *)lds_dst = *(const float4 *)gsrc;
#endif
}

__device__ __forceinline__ void async_copies_done()
{
#if defined(HAVE_ASYNC_LDS)
#if __has_builtin(__builtin_amdgcn_s_wait_asynccnt)
  __builtin_amdgcn_s_wait_asynccnt(0);
#else
  asm volatile("s_wait_asynccnt 0" ::: "memory");
#endif
#endif
}

// ---------------------------------------------------------------------------
// WMMA inner step for a 256x64x32 tile: wave w owns rows w*32..w*32+31
// (2 A-fragments), 4 B-fragments reused across both -> 8 WMMAs per call.
// A fragment (16-bit 16x32): lanes 0-15 hold K 0-7/16-23, lanes 16-31 hold
// K 8-15/24-31.  B staged [n][k] so a lane's 16 K-elems are two contiguous 8-runs (same K order as A).
// ---------------------------------------------------------------------------
template <int MF>
__device__ __forceinline__ void wmma_tiles(v8f (&acc)[MF][4],
                                           const bf16 (*As)[APAD],
                                           const bf16 (*Bs)[BPAD],
                                           int wave, int lane)
{
  v16bf a[MF];
#pragma unroll
  for (int mf = 0; mf < MF; ++mf) {
    const int mrow = wave * (16 * MF) + mf * 16 + (lane & 15);
    const int ka   = (lane >> 4) * 8;
    ((float4 *)&a[mf])[0] = *(const float4 *)&As[mrow][ka];
    ((float4 *)&a[mf])[1] = *(const float4 *)&As[mrow][ka + 16];
  }
#pragma unroll
  for (int t = 0; t < 4; ++t) {
    v16bf b;
    const int col = t * 16 + (lane & 15);
    const int kbq = (lane >> 4) * 8;
    ((float4 *)&b)[0] = *(const float4 *)&Bs[col][kbq];
    ((float4 *)&b)[1] = *(const float4 *)&Bs[col][kbq + 16];
#pragma unroll
    for (int mf = 0; mf < MF; ++mf)
      acc[mf][t] = __builtin_amdgcn_wmma_f32_16x16x32_bf16(
          false, a[mf], false, b, (short)0, acc[mf][t], false, false);
  }
}

enum GemmOut { OUT_F32 = 0, OUT_BF16 = 1, OUT_BF16_RELU = 2 };

// ---------------------------------------------------------------------------
// Generic GEMM: C[M x N] = A_bf16[M x K] * bf16(W_f32[K x N]) + bias.
// Output mode is a template parameter -> branch-free epilogue.
// Grid: (N/BN, M/BM), 256 threads.
// ---------------------------------------------------------------------------
template <int MODE>
__global__ __launch_bounds__(256) void
k_gemm_xw(const bf16 *__restrict__ A, const float *__restrict__ W,
          const float *__restrict__ bias, void *__restrict__ Cout,
          int N, int K)
{
  __shared__ bf16 As[BM][APAD];
  __shared__ bf16 Bs[BN][BPAD];
  const int tid = threadIdx.x, lane = tid & 31, wave = tid >> 5;
  const int m0 = blockIdx.y * BM, n0 = blockIdx.x * BN;
  v8f acc[2][4];
#pragma unroll
  for (int mf = 0; mf < 2; ++mf)
#pragma unroll
    for (int t = 0; t < 4; ++t)
#pragma unroll
      for (int j = 0; j < 8; ++j) acc[mf][t][j] = 0.0f;

  const int kk = tid >> 3, n8 = (tid & 7) * 8;   // B-stage coords
  for (int k0 = 0; k0 < K; k0 += BK) {
    __syncthreads();
#pragma unroll
    for (int it = 0; it < 4; ++it) {             // A: 256 rows x 32 k, 16B/thread
      int slot = tid + it * 256;
      int r = slot >> 2, c8 = (slot & 3) * 8;
      lds_cp16(&As[r][c8], A + (size_t)(m0 + r) * K + k0 + c8);
    }
    {                                            // B: 32k x 64n fp32->bf16, [n][k]
      const float *src = W + (size_t)(k0 + kk) * N + n0 + n8;
      float4 f0 = *(const float4 *)src;
      float4 f1 = *(const float4 *)(src + 4);
      Bs[n8 + 0][kk] = (bf16)f0.x; Bs[n8 + 1][kk] = (bf16)f0.y;
      Bs[n8 + 2][kk] = (bf16)f0.z; Bs[n8 + 3][kk] = (bf16)f0.w;
      Bs[n8 + 4][kk] = (bf16)f1.x; Bs[n8 + 5][kk] = (bf16)f1.y;
      Bs[n8 + 6][kk] = (bf16)f1.z; Bs[n8 + 7][kk] = (bf16)f1.w;
      if (k0 + BK < K)                           // stream next weight panel
        __builtin_prefetch(src + (size_t)BK * N, 0, 1);
    }
    async_copies_done();
    __syncthreads();
    wmma_tiles<2>(acc, As, Bs, wave, lane);
  }

  // C layout: VGPR j -> row j (lanes 0-15) / j+8 (lanes 16-31), col = lane%16
  float *Cf = (float *)Cout;
  bf16  *Cb = (bf16 *)Cout;
  const int c0 = n0 + (lane & 15);
#pragma unroll
  for (int mf = 0; mf < 2; ++mf) {
    const int r0 = m0 + wave * 32 + mf * 16 + (lane >> 4) * 8;
#pragma unroll
    for (int t = 0; t < 4; ++t) {
      const int c = c0 + t * 16;
      const float bb = bias[c];
#pragma unroll
      for (int j = 0; j < 8; ++j) {
        float v = acc[mf][t][j] + bb;
        if (MODE == OUT_BF16_RELU) v = fmaxf(v, 0.0f);
        size_t idx = (size_t)(r0 + j) * N + c;
        if (MODE == OUT_F32) Cf[idx] = v;
        else                 Cb[idx] = (bf16)v;
      }
    }
  }
}

// ---------------------------------------------------------------------------
// Attention scores: S[b,h] = Q[b,h] (T x dk) * K[b,h]^T, scaled.
// Grid: (T/BN, T/BM, B*H).  Both stages are pure copies -> async path.
// ---------------------------------------------------------------------------
__global__ __launch_bounds__(256) void
k_scores(const bf16 *__restrict__ Q, const bf16 *__restrict__ Kc,
         float *__restrict__ S, float scale)
{
  __shared__ bf16 As[BM][APAD];
  __shared__ bf16 Bs[BN][BPAD];
  const int tid = threadIdx.x, lane = tid & 31, wave = tid >> 5;
  const int m0 = blockIdx.y * BM, n0 = blockIdx.x * BN;
  const int bh = blockIdx.z, b = bh >> 4, h = bh & 15;
  const bf16 *Qb = Q  + (size_t)b * TSEQ * DM + (size_t)h * HDK;
  const bf16 *Kb = Kc + (size_t)b * TSEQ * DM + (size_t)h * HDK;
  v8f acc[2][4];
#pragma unroll
  for (int mf = 0; mf < 2; ++mf)
#pragma unroll
    for (int t = 0; t < 4; ++t)
#pragma unroll
      for (int j = 0; j < 8; ++j) acc[mf][t][j] = 0.0f;

  for (int k0 = 0; k0 < HDK; k0 += BK) {
    __syncthreads();
#pragma unroll
    for (int it = 0; it < 4; ++it) {
      int slot = tid + it * 256;
      int r = slot >> 2, c8 = (slot & 3) * 8;
      lds_cp16(&As[r][c8], Qb + (size_t)(m0 + r) * DM + k0 + c8);
    }
    {                                  // B = K^T: [n][k] stage is a straight copy
      int n = tid >> 2, c8 = (tid & 3) * 8;
      lds_cp16(&Bs[n][c8], Kb + (size_t)(n0 + n) * DM + k0 + c8);
    }
    async_copies_done();
    __syncthreads();
    wmma_tiles<2>(acc, As, Bs, wave, lane);
  }

  const int c0 = n0 + (lane & 15);
#pragma unroll
  for (int mf = 0; mf < 2; ++mf) {
    const int r0 = m0 + wave * 32 + mf * 16 + (lane >> 4) * 8;
#pragma unroll
    for (int t = 0; t < 4; ++t)
#pragma unroll
      for (int j = 0; j < 8; ++j)
        S[((size_t)bh * TSEQ + r0 + j) * TSEQ + c0 + t * 16] =
            acc[mf][t][j] * scale;
  }
}

// ---------------------------------------------------------------------------
// Context: O[b,h] = W (T x T, softmaxed bf16) * V[b,h] (T x dk).
// Grid: (1, T/BM, B*H).  Writes bf16 in concat-heads layout [b*T+t][h*64+n].
// ---------------------------------------------------------------------------
__global__ __launch_bounds__(256) void
k_ctx(const bf16 *__restrict__ Wt, const bf16 *__restrict__ Vc,
      bf16 *__restrict__ O)
{
  __shared__ bf16 As[BM][APAD];
  __shared__ bf16 Bs[BN][BPAD];
  const int tid = threadIdx.x, lane = tid & 31, wave = tid >> 5;
  const int m0 = blockIdx.y * BM;
  const int bh = blockIdx.z, b = bh >> 4, h = bh & 15;
  const bf16 *Ab = Wt + (size_t)bh * TSEQ * TSEQ;
  const bf16 *Vb = Vc + (size_t)b * TSEQ * DM + (size_t)h * HDK;
  v8f acc[2][4];
#pragma unroll
  for (int mf = 0; mf < 2; ++mf)
#pragma unroll
    for (int t = 0; t < 4; ++t)
#pragma unroll
      for (int j = 0; j < 8; ++j) acc[mf][t][j] = 0.0f;

  const int kk = tid >> 3, n8 = (tid & 7) * 8;
  for (int k0 = 0; k0 < TSEQ; k0 += BK) {
    __syncthreads();
#pragma unroll
    for (int it = 0; it < 4; ++it) {
      int slot = tid + it * 256;
      int r = slot >> 2, c8 = (slot & 3) * 8;
      lds_cp16(&As[r][c8], Ab + (size_t)(m0 + r) * TSEQ + k0 + c8);
    }
    {                                            // V rows -> Bs[n][k] transpose
      const bf16 *src = Vb + (size_t)(k0 + kk) * DM + n8;
#pragma unroll
      for (int j = 0; j < 8; ++j) Bs[n8 + j][kk] = src[j];
    }
    async_copies_done();
    __syncthreads();
    wmma_tiles<2>(acc, As, Bs, wave, lane);
  }

  const int c0 = lane & 15;
#pragma unroll
  for (int mf = 0; mf < 2; ++mf) {
    const int r0 = m0 + wave * 32 + mf * 16 + (lane >> 4) * 8;
#pragma unroll
    for (int t = 0; t < 4; ++t)
#pragma unroll
      for (int j = 0; j < 8; ++j)
        O[(size_t)(b * TSEQ + r0 + j) * DM + h * HDK + c0 + t * 16] =
            (bf16)acc[mf][t][j];
  }
}

// ---------------------------------------------------------------------------
// Row softmax, in place on the fp32 d_out slice; also emits bf16 for k_ctx.
// causal=1 masks k > q (tril).  One 128-thread block per row.
// ---------------------------------------------------------------------------
__global__ __launch_bounds__(128) void
k_softmax(float *__restrict__ S, bf16 *__restrict__ Wb, int causal)
{
  __shared__ float red[128];
  const int row = blockIdx.x;               // (b*H+h)*T + q
  const int q = row & (TSEQ - 1);
  float *p  = S  + (size_t)row * TSEQ;
  bf16  *pw = Wb + (size_t)row * TSEQ;
  const int limit = causal ? (q + 1) : TSEQ;

  float v[4];
  float mx = -3.0e38f;
#pragma unroll
  for (int j = 0; j < 4; ++j) {
    int c = threadIdx.x + j * 128;
    v[j] = (c < limit) ? p[c] : -3.0e38f;
    mx = fmaxf(mx, v[j]);
  }
  red[threadIdx.x] = mx;
  __syncthreads();
  for (int s = 64; s > 0; s >>= 1) {
    if (threadIdx.x < s)
      red[threadIdx.x] = fmaxf(red[threadIdx.x], red[threadIdx.x + s]);
    __syncthreads();
  }
  mx = red[0];
  __syncthreads();

  float sum = 0.0f;
#pragma unroll
  for (int j = 0; j < 4; ++j) {
    int c = threadIdx.x + j * 128;
    v[j] = (c < limit) ? __expf(v[j] - mx) : 0.0f;
    sum += v[j];
  }
  red[threadIdx.x] = sum;
  __syncthreads();
  for (int s = 64; s > 0; s >>= 1) {
    if (threadIdx.x < s) red[threadIdx.x] += red[threadIdx.x + s];
    __syncthreads();
  }
  const float inv = 1.0f / red[0];
#pragma unroll
  for (int j = 0; j < 4; ++j) {
    int c = threadIdx.x + j * 128;
    float o = v[j] * inv;
    p[c]  = o;
    pw[c] = (bf16)o;
  }
}

// ---------------------------------------------------------------------------
// x = LayerNorm(x + a) * g + b; emits fp32 (residual stream) + bf16 copy.
// ---------------------------------------------------------------------------
__global__ __launch_bounds__(256) void
k_add_ln(const float *__restrict__ X, const float *__restrict__ Aa,
         const float *__restrict__ g, const float *__restrict__ bta,
         float *__restrict__ Xo, bf16 *__restrict__ Xb)
{
  __shared__ float red[256];
  const int row = blockIdx.x;
  float v[4];
  float s = 0.0f;
#pragma unroll
  for (int j = 0; j < 4; ++j) {
    int c = threadIdx.x + j * 256;
    size_t idx = (size_t)row * DM + c;
    v[j] = X[idx] + Aa[idx];
    s += v[j];
  }
  red[threadIdx.x] = s;
  __syncthreads();
  for (int st = 128; st > 0; st >>= 1) {
    if (threadIdx.x < st) red[threadIdx.x] += red[threadIdx.x + st];
    __syncthreads();
  }
  const float mean = red[0] * (1.0f / DM);
  __syncthreads();

  float s2 = 0.0f;
#pragma unroll
  for (int j = 0; j < 4; ++j) { float d = v[j] - mean; s2 += d * d; }
  red[threadIdx.x] = s2;
  __syncthreads();
  for (int st = 128; st > 0; st >>= 1) {
    if (threadIdx.x < st) red[threadIdx.x] += red[threadIdx.x + st];
    __syncthreads();
  }
  const float inv = rsqrtf(red[0] * (1.0f / DM) + 1e-5f);
#pragma unroll
  for (int j = 0; j < 4; ++j) {
    int c = threadIdx.x + j * 256;
    size_t idx = (size_t)row * DM + c;
    float y = g[c] * (v[j] - mean) * inv + bta[c];
    Xo[idx] = y;
    Xb[idx] = (bf16)y;
  }
}

// ---------------------------------------------------------------------------
// x = emb[tgt] * sqrt(D) + positional encoding.  One block per token.
// ---------------------------------------------------------------------------
__global__ __launch_bounds__(256) void
k_embed(const int *__restrict__ tgt, const float *__restrict__ E,
        float *__restrict__ Xo, bf16 *__restrict__ Xb)
{
  const int m = blockIdx.x;
  const int tok = tgt[m];
  const int pos = m & (TSEQ - 1);
#pragma unroll
  for (int j = 0; j < 4; ++j) {
    int c = threadIdx.x + j * 256;
    float freq = __expf((float)(c & ~1) * (-9.210340372f / (float)DM));
    float ang = (float)pos * freq;
    float pe = (c & 1) ? __cosf(ang) : __sinf(ang);
    float val = E[(size_t)tok * DM + c] * 32.0f + pe;   // sqrt(1024)=32
    size_t idx = (size_t)m * DM + c;
    Xo[idx] = val;
    Xb[idx] = (bf16)val;
  }
}

__global__ __launch_bounds__(256) void
k_f32_to_bf16(const float *__restrict__ in, bf16 *__restrict__ out, size_t n)
{
  size_t i = ((size_t)blockIdx.x * 256 + threadIdx.x) * 4;
  if (i + 3 < n) {
    float4 f = *(const float4 *)(in + i);
    out[i + 0] = (bf16)f.x; out[i + 1] = (bf16)f.y;
    out[i + 2] = (bf16)f.z; out[i + 3] = (bf16)f.w;
  }
}

// ---------------------------------------------------------------------------
extern "C" void kernel_launch(void *const *d_in, const int *in_sizes, int n_in,
                              void *d_out, int out_size, void *d_ws,
                              size_t ws_size, hipStream_t stream)
{
  (void)in_sizes; (void)n_in; (void)out_size; (void)ws_size;
  const int   *tgt  = (const int *)  d_in[0];
  const float *enc  = (const float *)d_in[1];
  /* d_in[2] = tgt_mask (causality applied analytically) */
  const float *embw = (const float *)d_in[3];
  const float *Wout = (const float *)d_in[4];
  const float *bout = (const float *)d_in[5];
  const float *Wq_s = (const float *)d_in[6],  *bq_s = (const float *)d_in[7];
  const float *Wk_s = (const float *)d_in[8],  *bk_s = (const float *)d_in[9];
  const float *Wv_s = (const float *)d_in[10], *bv_s = (const float *)d_in[11];
  const float *Wo_s = (const float *)d_in[12], *bo_s = (const float *)d_in[13];
  const float *Wq_c = (const float *)d_in[14], *bq_c = (const float *)d_in[15];
  const float *Wk_c = (const float *)d_in[16], *bk_c = (const float *)d_in[17];
  const float *Wv_c = (const float *)d_in[18], *bv_c = (const float *)d_in[19];
  const float *Wo_c = (const float *)d_in[20], *bo_c = (const float *)d_in[21];
  const float *W1   = (const float *)d_in[22], *b1   = (const float *)d_in[23];
  const float *W2   = (const float *)d_in[24], *b2   = (const float *)d_in[25];
  const float *ln1g = (const float *)d_in[26], *ln1b = (const float *)d_in[27];
  const float *ln2g = (const float *)d_in[28], *ln2b = (const float *)d_in[29];
  const float *ln3g = (const float *)d_in[30], *ln3b = (const float *)d_in[31];

  // ---- workspace carve-up (~92 MB) ----
  char *p = (char *)d_ws;
  auto alloc = [&](size_t bytes) {
    char *r = p;
    p += (bytes + 255) & ~(size_t)255;
    return (void *)r;
  };
  float *xf   = (float *)alloc((size_t)MTOK * DM * 4);
  float *tmp  = (float *)alloc((size_t)MTOK * DM * 4);
  bf16  *xb   = (bf16 *) alloc((size_t)MTOK * DM * 2);
  bf16  *encb = (bf16 *) alloc((size_t)MTOK * DM * 2);
  bf16  *qb   = (bf16 *) alloc((size_t)MTOK * DM * 2);
  bf16  *kb   = (bf16 *) alloc((size_t)MTOK * DM * 2);
  bf16  *vb   = (bf16 *) alloc((size_t)MTOK * DM * 2);
  bf16  *cxb  = (bf16 *) alloc((size_t)MTOK * DM * 2);
  bf16  *f1b  = (bf16 *) alloc((size_t)MTOK * DFF_N * 2);
  bf16  *wbuf = (bf16 *) alloc((size_t)NB * NH * TSEQ * TSEQ * 2);

  float *logits = (float *)d_out;
  const size_t wsl = (size_t)NB * NH * TSEQ * TSEQ;  // per-layer weight slice
  float *selfw  = logits + (size_t)MTOK * NVOC;
  float *crossw = selfw + (size_t)NL * wsl;

  const dim3 blk(256);
  const dim3 gP (DM    / BN, MTOK / BM);
  const dim3 gF1(DFF_N / BN, MTOK / BM);
  const dim3 gLg(NVOC  / BN, MTOK / BM);
  const dim3 gS (TSEQ / BN, TSEQ / BM, NB * NH);
  const dim3 gC (1,          TSEQ / BM, NB * NH);
  const float scale = 0.125f;               // 1/sqrt(64)

  k_embed<<<MTOK, blk, 0, stream>>>(tgt, embw, xf, xb);
  {
    size_t n = (size_t)MTOK * DM;
    k_f32_to_bf16<<<(unsigned)(n / 4 / 256), blk, 0, stream>>>(enc, encb, n);
  }

  for (int i = 0; i < NL; ++i) {
    const size_t oWW = (size_t)i * DM * DM, oB = (size_t)i * DM;
    float *sw = selfw  + (size_t)i * wsl;
    float *cw = crossw + (size_t)i * wsl;

    // ---- self attention ----
    k_gemm_xw<OUT_BF16><<<gP, blk, 0, stream>>>(xb, Wq_s + oWW, bq_s + oB, qb, DM, DM);
    k_gemm_xw<OUT_BF16><<<gP, blk, 0, stream>>>(xb, Wk_s + oWW, bk_s + oB, kb, DM, DM);
    k_gemm_xw<OUT_BF16><<<gP, blk, 0, stream>>>(xb, Wv_s + oWW, bv_s + oB, vb, DM, DM);
    k_scores <<<gS, blk, 0, stream>>>(qb, kb, sw, scale);
    k_softmax<<<NB * NH * TSEQ, dim3(128), 0, stream>>>(sw, wbuf, 1);
    k_ctx    <<<gC, blk, 0, stream>>>(wbuf, vb, cxb);
    k_gemm_xw<OUT_F32><<<gP, blk, 0, stream>>>(cxb, Wo_s + oWW, bo_s + oB, tmp, DM, DM);
    k_add_ln <<<MTOK, blk, 0, stream>>>(xf, tmp, ln1g + oB, ln1b + oB, xf, xb);

    // ---- cross attention ----
    k_gemm_xw<OUT_BF16><<<gP, blk, 0, stream>>>(xb,   Wq_c + oWW, bq_c + oB, qb, DM, DM);
    k_gemm_xw<OUT_BF16><<<gP, blk, 0, stream>>>(encb, Wk_c + oWW, bk_c + oB, kb, DM, DM);
    k_gemm_xw<OUT_BF16><<<gP, blk, 0, stream>>>(encb, Wv_c + oWW, bv_c + oB, vb, DM, DM);
    k_scores <<<gS, blk, 0, stream>>>(qb, kb, cw, scale);
    k_softmax<<<NB * NH * TSEQ, dim3(128), 0, stream>>>(cw, wbuf, 0);
    k_ctx    <<<gC, blk, 0, stream>>>(wbuf, vb, cxb);
    k_gemm_xw<OUT_F32><<<gP, blk, 0, stream>>>(cxb, Wo_c + oWW, bo_c + oB, tmp, DM, DM);
    k_add_ln <<<MTOK, blk, 0, stream>>>(xf, tmp, ln2g + oB, ln2b + oB, xf, xb);

    // ---- FFN ----
    k_gemm_xw<OUT_BF16_RELU><<<gF1, blk, 0, stream>>>(
        xb, W1 + (size_t)i * DM * DFF_N, b1 + (size_t)i * DFF_N, f1b, DFF_N, DM);
    k_gemm_xw<OUT_F32><<<gP, blk, 0, stream>>>(
        f1b, W2 + (size_t)i * DFF_N * DM, b2 + oB, tmp, DM, DFF_N);
    k_add_ln <<<MTOK, blk, 0, stream>>>(xf, tmp, ln3g + oB, ln3b + oB, xf, xb);
  }

  // ---- final logits ----
  k_gemm_xw<OUT_F32><<<gLg, blk, 0, stream>>>(xb, Wout, bout, logits, NVOC, DM);
}
